// ScaledDotAttention_79714593013963
// MI455X (gfx1250) — matrix operand order repaired and checked by reading, not compiled, not run on hardware
//
#include <hip/hip_runtime.h>

typedef unsigned short u16;
typedef unsigned int   u32;

typedef __attribute__((ext_vector_type(16))) __bf16 v16bf;
typedef __attribute__((ext_vector_type(8)))  float  v8f;

#define N_HEADS 16
#define DIM     1024
#define DPH     64
#define BS      2
#define SLEN    2048
#define ROWS    (BS*SLEN)

union FragBF { v16bf v; u32 u[8]; u16 h[16]; };
union FragF  { v8f   v; float f[8]; };

__device__ __forceinline__ u16 f2bf(float f) {
  u32 u = __float_as_uint(f);
  u32 r = u + 0x7FFFu + ((u >> 16) & 1u);   // round-to-nearest-even
  return (u16)(r >> 16);
}

// reductions restricted to 16-lane halves of the wave32 (xor masks < 16)
__device__ __forceinline__ float hmax16(float x) {
#pragma unroll
  for (int off = 8; off > 0; off >>= 1) x = fmaxf(x, __shfl_xor(x, off, 32));
  return x;
}
__device__ __forceinline__ float hsum16(float x) {
#pragma unroll
  for (int off = 8; off > 0; off >>= 1) x += __shfl_xor(x, off, 32);
  return x;
}

__device__ __forceinline__ v8f wmma_bf16(const FragBF& a, const FragBF& b, v8f c) {
  // (neg_a, A, neg_b, B, c_mod, C, reuse_a, reuse_b)
  return __builtin_amdgcn_wmma_f32_16x16x32_bf16(false, a.v, false, b.v,
                                                 (short)0, c, false, false);
}

// async global -> LDS copy of 8 bytes (per-lane), tracked by ASYNCcnt
__device__ __forceinline__ void async_g2l_b64(void* lds_ptr, const void* gptr) {
  u32 lds_off = (u32)(unsigned long long)lds_ptr;  // LDS offset lives in addr[31:0]
  asm volatile("global_load_async_to_lds_b64 %0, %1, off"
               :: "v"(lds_off), "v"(gptr) : "memory");
}
__device__ __forceinline__ void wait_async0() {
  asm volatile("s_wait_asynccnt 0" ::: "memory");
}

// ---------------------------------------------------------------- f32 -> bf16
__global__ void cvt_bf16_kernel(const float* __restrict__ src,
                                u16* __restrict__ dst, int n4) {
  int i = blockIdx.x * blockDim.x + threadIdx.x;
  if (i < n4) {
    float4 f = ((const float4*)src)[i];
    u32 lo = (u32)f2bf(f.x) | ((u32)f2bf(f.y) << 16);
    u32 hi = (u32)f2bf(f.z) | ((u32)f2bf(f.w) << 16);
    ((uint2*)dst)[i] = make_uint2(lo, hi);
  }
}

// ---------------------------------------------------------------- GEMM
// C = A(bf16, ROWSxDIM) * W(bf16, DIMxDIM) + bias
// MODE 0: V  -> bias, store bf16 TRANSPOSED per batch:  vt[b][col][s]
// MODE 1: K  -> bias, L2-normalize per 64-col head, store bf16
// MODE 2: Q  -> bias, normalize, * attention_scale, store bf16
// MODE 3: O  -> bias, store f32 (final output)
#define GEMM_BM 512
#define GEMM_BN 64
#define GEMM_BK 32
#define XS_STRIDE 36   // 32 bf16 + pad (bank spread, keeps 8B alignment)
#define WS_STRIDE 68   // 64 bf16 + pad

template <int MODE>
__global__ void __launch_bounds__(256)
gemm_kernel(const u16* __restrict__ A, const u16* __restrict__ W,
            const float* __restrict__ bias, u16* __restrict__ outb,
            float* __restrict__ outf, const float* __restrict__ scale_ptr) {
  __shared__ u16 Xs[GEMM_BM * XS_STRIDE];
  __shared__ u16 Ws[GEMM_BK * WS_STRIDE];

  const int tid = threadIdx.x;
  const int wv  = tid >> 5;
  const int ln  = tid & 31;
  const int n   = ln & 15;
  const int hh  = ln >> 4;                 // lane half: 0/1
  const int blockRow = blockIdx.x * GEMM_BM;
  const int colbase  = blockIdx.y * GEMM_BN;
  const int wrow     = wv * 64;            // wave's 64-row slice inside block

  FragF acc[4][4];
#pragma unroll
  for (int i = 0; i < 4; i++)
#pragma unroll
    for (int j = 0; j < 4; j++)
#pragma unroll
      for (int e = 0; e < 8; e++) acc[i][j].f[e] = 0.f;

  for (int k0 = 0; k0 < DIM; k0 += GEMM_BK) {
    __syncthreads();   // previous slab fully consumed before DMA overwrites it
    // stage X slab: 512 x 32 bf16 via async DMA (8B per op, no VGPR roundtrip)
#pragma unroll
    for (int t = 0; t < 16; t++) {
      int idx = tid + t * 256;             // 0..4095
      int r   = idx >> 3;
      int c4  = (idx & 7) * 4;
      async_g2l_b64(Xs + r * XS_STRIDE + c4,
                    A + (size_t)(blockRow + r) * DIM + k0 + c4);
    }
    // stage W slab: 32 x 64 bf16
#pragma unroll
    for (int t = 0; t < 2; t++) {
      int idx = tid + t * 256;             // 0..511
      int kr  = idx >> 4;
      int c4  = (idx & 15) * 4;
      async_g2l_b64(Ws + kr * WS_STRIDE + c4,
                    W + (size_t)(k0 + kr) * DIM + colbase + c4);
    }
    // warm L2/WGP$ for the next slab while this one computes
    if (k0 + GEMM_BK < DIM) {
      __builtin_prefetch(A + (size_t)(blockRow + 2 * tid) * DIM + k0 + GEMM_BK, 0, 1);
      __builtin_prefetch(A + (size_t)(blockRow + 2 * tid + 1) * DIM + k0 + GEMM_BK, 0, 1);
      if (tid < GEMM_BK)
        __builtin_prefetch(W + (size_t)(k0 + GEMM_BK + tid) * DIM + colbase, 0, 1);
    }
    wait_async0();     // this wave's DMA landed in LDS
    __syncthreads();   // everyone's DMA landed

    // A fragments: 16x32 bf16; lane = row, K 0-15 lanes 0-15 / 16-31 lanes 16-31
    FragBF af[4], bf[4];
    const int kbA = hh * 16;
#pragma unroll
    for (int ri = 0; ri < 4; ri++) {
      const u16* base = Xs + (wrow + ri * 16 + n) * XS_STRIDE;
#pragma unroll
      for (int j = 0; j < 8; j++) {
        af[ri].u[j]     = *(const u32*)(base + kbA + 2 * j);
      }
    }
    // B fragments: 32x16 bf16; lane = col, K 0-15 lanes 0-15 / 16-31 lanes 16-31
    const int kbB = hh * 16;
#pragma unroll
    for (int ci = 0; ci < 4; ci++) {
#pragma unroll
      for (int v = 0; v < 8; v++) {
        u32 e0 = Ws[(kbB + 2 * v)     * WS_STRIDE + ci * 16 + n];
        u32 e1 = Ws[(kbB + 2 * v + 1) * WS_STRIDE + ci * 16 + n];
        bf[ci].u[v] = e0 | (e1 << 16);
      }
    }
#pragma unroll
    for (int ri = 0; ri < 4; ri++)
#pragma unroll
      for (int ci = 0; ci < 4; ci++)
        acc[ri][ci].v = wmma_bf16(af[ri], bf[ci], acc[ri][ci].v);
  }

  // ---- epilogue: bias
  float bcol[4];
#pragma unroll
  for (int ci = 0; ci < 4; ci++) bcol[ci] = bias[colbase + ci * 16 + n];
#pragma unroll
  for (int ri = 0; ri < 4; ri++)
#pragma unroll
    for (int ci = 0; ci < 4; ci++)
#pragma unroll
      for (int r = 0; r < 8; r++) acc[ri][ci].f[r] += bcol[ci];

  // ---- fused L2 row-normalization over this wave's 64 cols (== one head)
  if (MODE == 1 || MODE == 2) {
    float asc = 1.0f;
    if (MODE == 2) asc = scale_ptr[0];
#pragma unroll
    for (int ri = 0; ri < 4; ri++) {
#pragma unroll
      for (int r = 0; r < 8; r++) {
        float s = 0.f;
#pragma unroll
        for (int ci = 0; ci < 4; ci++) {
          float t = acc[ri][ci].f[r];
          s += t * t;
        }
        s = hsum16(s);                          // full-row sum-of-squares
        float rn = asc / fmaxf(sqrtf(s), 1e-12f);
#pragma unroll
        for (int ci = 0; ci < 4; ci++) acc[ri][ci].f[r] *= rn;
      }
    }
  }

  // ---- store (C layout: VGPR r -> row r or r+8 by lane half; col = lane&15)
#pragma unroll
  for (int ri = 0; ri < 4; ri++) {
#pragma unroll
    for (int r = 0; r < 8; r++) {
      int grow = blockRow + wrow + ri * 16 + (hh ? r + 8 : r);
#pragma unroll
      for (int ci = 0; ci < 4; ci++) {
        int col   = colbase + ci * 16 + n;
        float val = acc[ri][ci].f[r];
        if (MODE == 3) {
          outf[(size_t)grow * DIM + col] = val;
        } else if (MODE == 0) {             // V: transposed for P@V dword loads
          int bb = grow >> 11, ss = grow & (SLEN - 1);
          outb[((size_t)bb * DIM + col) * SLEN + ss] = f2bf(val);
        } else {
          outb[(size_t)grow * DIM + col] = f2bf(val);
        }
      }
    }
  }
}

// ---------------------------------------------------------------- attention
// One wave handles 16 q-rows of one (b,h); online softmax over 32-key steps.
#define PS_STRIDE 34

__global__ void __launch_bounds__(128)
attn_kernel(const u16* __restrict__ qn, const u16* __restrict__ kn,
            const u16* __restrict__ vt, u16* __restrict__ ctx) {
  __shared__ u16 Ps[4 * 16 * PS_STRIDE];   // per-wave P transpose scratch

  const int tid = threadIdx.x;
  const int wv  = tid >> 5;
  const int ln  = tid & 31;
  const int n   = ln & 15;
  const int hh  = ln >> 4;
  const int b   = blockIdx.y / N_HEADS;
  const int h   = blockIdx.y % N_HEADS;
  const int q0  = blockIdx.x * 64 + wv * 16;

  u16* ps = Ps + wv * 16 * PS_STRIDE;

  // Q fragments (A-layout, d-chunks 0-31 / 32-63), resident for whole loop
  FragBF qa[2];
  {
    const u16* qrow = qn + (size_t)(b * SLEN + q0 + n) * DIM + h * DPH;
    const int kb = hh * 16;
#pragma unroll
    for (int c = 0; c < 2; c++)
#pragma unroll
      for (int j = 0; j < 8; j++) {
        qa[c].u[j]     = *(const u32*)(qrow + c * 32 + kb + 2 * j);
      }
  }

  float mrow[8], lrow[8];
  FragF acc[4];
#pragma unroll
  for (int r = 0; r < 8; r++) { mrow[r] = -1e30f; lrow[r] = 0.f; }
#pragma unroll
  for (int ci = 0; ci < 4; ci++)
#pragma unroll
    for (int r = 0; r < 8; r++) acc[ci].f[r] = 0.f;

  const int nk = (q0 + 16 + 31) & ~31;     // causal key extent, 32-aligned

  for (int kb0 = 0; kb0 < nk; kb0 += 32) {
    // ---- scores: two 16x16 tiles, K(contract)=64 -> 2 WMMAs each
    FragF S[2];
#pragma unroll
    for (int t = 0; t < 2; t++) {
      const u16* krow =
          kn + (size_t)(b * SLEN + kb0 + t * 16 + n) * DIM + h * DPH + hh * 16;
      FragBF kf0, kf1;
#pragma unroll
      for (int v = 0; v < 8; v++) {
        kf0.u[v] = *(const u32*)(krow + 2 * v);        // d in [0,32)
        kf1.u[v] = *(const u32*)(krow + 32 + 2 * v);   // d in [32,64)
      }
      FragF z;
#pragma unroll
      for (int e = 0; e < 8; e++) z.f[e] = 0.f;
      S[t].v = wmma_bf16(qa[1], kf1, wmma_bf16(qa[0], kf0, z.v));
    }

    // ---- causal mask + online softmax (per row, per 16-lane half)
#pragma unroll
    for (int r = 0; r < 8; r++) {
      int row = hh ? r + 8 : r;
      int qr  = q0 + row;
      if (kb0 + n      > qr) S[0].f[r] = -1e30f;
      if (kb0 + 16 + n > qr) S[1].f[r] = -1e30f;
      float t    = hmax16(fmaxf(S[0].f[r], S[1].f[r]));
      float mnew = fmaxf(mrow[r], t);
      float corr = __expf(mrow[r] - mnew);
      float p0   = __expf(S[0].f[r] - mnew);
      float p1   = __expf(S[1].f[r] - mnew);
      lrow[r] = lrow[r] * corr + hsum16(p0 + p1);
      mrow[r] = mnew;
#pragma unroll
      for (int ci = 0; ci < 4; ci++) acc[ci].f[r] *= corr;
      // C-layout -> LDS (bf16) for the A-layout reload
      ps[row * PS_STRIDE + n]      = f2bf(p0);
      ps[row * PS_STRIDE + 16 + n] = f2bf(p1);
    }
    // cross-lane LDS dependency within the wave: drain DScnt explicitly
    asm volatile("s_wait_dscnt 0" ::: "memory");

    // ---- P as A-fragment (16 q-rows x 32 keys)
    FragBF pa;
    {
      const u16* pr = ps + n * PS_STRIDE;
      const int kb = hh * 16;
#pragma unroll
      for (int j = 0; j < 8; j++) {
        pa.u[j]     = *(const u32*)(pr + kb + 2 * j);
      }
    }
    // ---- P @ V  (V stored transposed: key pairs contiguous -> dword loads)
#pragma unroll
    for (int ci = 0; ci < 4; ci++) {
      const u16* vcol =
          vt + ((size_t)b * DIM + h * DPH + ci * 16 + n) * SLEN + kb0 + hh * 16;
      FragBF vb;
#pragma unroll
      for (int v = 0; v < 8; v++) vb.u[v] = *(const u32*)(vcol + 2 * v);
      acc[ci].v = wmma_bf16(pa, vb, acc[ci].v);
    }
    asm volatile("" ::: "memory");
  }

  // ---- finalize: divide by softmax denom, emit bf16 ctx
#pragma unroll
  for (int r = 0; r < 8; r++) {
    float inv = 1.0f / lrow[r];
    int qr    = q0 + (hh ? r + 8 : r);
    u16* orow = ctx + (size_t)(b * SLEN + qr) * DIM + h * DPH;
#pragma unroll
    for (int ci = 0; ci < 4; ci++)
      orow[ci * 16 + n] = f2bf(acc[ci].f[r] * inv);
  }
}

// ---------------------------------------------------------------- launch
extern "C" void kernel_launch(void* const* d_in, const int* in_sizes, int n_in,
                              void* d_out, int out_size, void* d_ws,
                              size_t ws_size, hipStream_t stream) {
  const float* x   = (const float*)d_in[0];
  // d_in[1] = mask (causal; applied analytically in-kernel)
  const float* Wq  = (const float*)d_in[2];
  const float* bq  = (const float*)d_in[3];
  const float* Wk  = (const float*)d_in[4];
  const float* bk  = (const float*)d_in[5];
  const float* Wv  = (const float*)d_in[6];
  const float* bv  = (const float*)d_in[7];
  const float* Wo  = (const float*)d_in[8];
  const float* bo  = (const float*)d_in[9];
  const float* asc = (const float*)d_in[10];
  float* out = (float*)d_out;

  // workspace: bf16 copies + activation buffers (~48 MiB total)
  u16* xb  = (u16*)d_ws;
  u16* wqb = xb  + (size_t)ROWS * DIM;
  u16* wkb = wqb + (size_t)DIM * DIM;
  u16* wvb = wkb + (size_t)DIM * DIM;
  u16* wob = wvb + (size_t)DIM * DIM;
  u16* qnb = wob + (size_t)DIM * DIM;   // normalized*scale Q, bf16
  u16* knb = qnb + (size_t)ROWS * DIM;  // normalized K, bf16
  u16* vtb = knb + (size_t)ROWS * DIM;  // V transposed per batch, bf16
  u16* ctx = vtb + (size_t)ROWS * DIM;  // attention output, bf16

  const int n_x = ROWS * DIM / 4;
  const int n_w = DIM * DIM / 4;
  cvt_bf16_kernel<<<(n_x + 255) / 256, 256, 0, stream>>>(x,  xb,  n_x);
  cvt_bf16_kernel<<<(n_w + 255) / 256, 256, 0, stream>>>(Wq, wqb, n_w);
  cvt_bf16_kernel<<<(n_w + 255) / 256, 256, 0, stream>>>(Wk, wkb, n_w);
  cvt_bf16_kernel<<<(n_w + 255) / 256, 256, 0, stream>>>(Wv, wvb, n_w);
  cvt_bf16_kernel<<<(n_w + 255) / 256, 256, 0, stream>>>(Wo, wob, n_w);

  dim3 gg(ROWS / GEMM_BM, DIM / GEMM_BN);
  gemm_kernel<2><<<gg, 256, 0, stream>>>(xb, wqb, bq, qnb, nullptr, asc);
  gemm_kernel<1><<<gg, 256, 0, stream>>>(xb, wkb, bk, knb, nullptr, nullptr);
  gemm_kernel<0><<<gg, 256, 0, stream>>>(xb, wvb, bv, vtb, nullptr, nullptr);

  attn_kernel<<<dim3(SLEN / 64, BS * N_HEADS), 128, 0, stream>>>(qnb, knb, vtb,
                                                                 ctx);

  gemm_kernel<3><<<gg, 256, 0, stream>>>(ctx, wob, bo, nullptr, out, nullptr);
}
